// HierarchicalGNNModule_12524124635271
// MI455X (gfx1250) — compile-verified, matrix operand order checked by reading, not run on hardware
//
#include <hip/hip_runtime.h>
#include <hip/hip_bf16.h>

// ---------------------------------------------------------------------------
// HierarchicalGNNModule on MI455X (gfx1250, wave32).
// Heavy math: v_wmma_f32_16x16x32_bf16 (fp32 accumulate).
// GEMM is templated on operand layout so inner loops are branch-free and
// fragment loads are 16-byte vector loads where the layout is contiguous.
// Attention: flash-style online softmax with soft adjacency mask; row
// reductions use DPP16 ROW_XMASK butterflies (pure VALU, no LDS waits);
// P staged through LDS (s_wait_dscnt) to re-layout C-operand -> A-operand.
// __launch_bounds__(32,1) gives the full VGPR budget -> no scratch spills.
// Params are consumed in setup_inputs() dict insertion order.
// ---------------------------------------------------------------------------

typedef __attribute__((ext_vector_type(16))) __bf16 v16bf;
typedef __attribute__((ext_vector_type(8)))  __bf16 v8bf;
typedef __attribute__((ext_vector_type(8)))  float  v8f;

#define ND    64
#define NH    8
#define NDH   512
#define NSEG  4096
#define NLOC  512
#define NREG  64

__device__ __forceinline__ v8f wmma_bf16(v16bf a, v16bf b, v8f c) {
  return __builtin_amdgcn_wmma_f32_16x16x32_bf16(false, a, false, b, (short)0, c,
                                                 false, false);
}

__device__ __forceinline__ v16bf cat8(v8bf lo, v8bf hi) {
  return __builtin_shufflevector(lo, hi, 0, 1, 2, 3, 4, 5, 6, 7,
                                 8, 9, 10, 11, 12, 13, 14, 15);
}

// DPP16 ROW_XMASK butterfly step (stays within a 16-lane row half).
template <int MASK>
__device__ __forceinline__ float row_xor(float v) {
  int i = __float_as_int(v);
  i = __builtin_amdgcn_update_dpp(i, i, 0x160 | MASK, 0xf, 0xf, true);
  return __int_as_float(i);
}
__device__ __forceinline__ float rmax16(float v) {
  v = fmaxf(v, row_xor<1>(v));
  v = fmaxf(v, row_xor<2>(v));
  v = fmaxf(v, row_xor<4>(v));
  v = fmaxf(v, row_xor<8>(v));
  return v;
}
__device__ __forceinline__ float rsum16(float v) {
  v += row_xor<1>(v);
  v += row_xor<2>(v);
  v += row_xor<4>(v);
  v += row_xor<8>(v);
  return v;
}

// fp32 -> bf16 convert
__global__ void k_f2b(const float* __restrict__ in, __bf16* __restrict__ out, int n) {
  int i = blockIdx.x * blockDim.x + threadIdx.x;
  if (i < n) out[i] = (__bf16)in[i];
}

// out = a + b (fp32 + bf16 mirror)
__global__ void k_addf(const float* __restrict__ a, const float* __restrict__ b,
                       float* __restrict__ of, __bf16* __restrict__ ob, int n) {
  int i = blockIdx.x * blockDim.x + threadIdx.x;
  if (i < n) {
    float v = a[i] + b[i];
    of[i] = v;
    if (ob) ob[i] = (__bf16)v;
  }
}

// ---------------------------------------------------------------------------
// bf16 WMMA GEMM: C[M,N] = act(op(A)[M,K] x op(B)[K,N] + bias)
// One wave computes a 64x16 strip (4 M-tiles share one B fragment).
// Template params: BF32 = B operand is fp32 (converted on the fly, used to
// stream seg_adj without a 32MB copy); TRA/TRB = operand stored transposed.
// Contiguous-K operands (A non-trans, B trans) use 16B vector loads.
// Requires M%64==0, N%16==0, K%32==0 (all dims here satisfy this).
// ---------------------------------------------------------------------------
template <bool BF32, bool TRA, bool TRB>
__global__ __launch_bounds__(32, 1)
void k_gemm_t(const __bf16* __restrict__ A, const void* __restrict__ Bv,
              const float* __restrict__ bias,
              float* __restrict__ Cf, __bf16* __restrict__ Cb,
              int M, int N, int K, int lda, int ldb, int relu) {
  const __bf16* Bh = (const __bf16*)Bv;
  const float*  Bf = (const float*)Bv;
  const int tn   = blockIdx.x * 16;
  const int tm0  = blockIdx.y * 64;
  const int lane = threadIdx.x;
  const int hgrp = lane >> 4;    // 0/1
  const int l16  = lane & 15;

  v8f acc[4];
#pragma unroll
  for (int t = 0; t < 4; ++t)
#pragma unroll
    for (int r = 0; r < 8; ++r) acc[t][r] = 0.0f;

  for (int k0 = 0; k0 < K; k0 += 32) {
    v16bf bfrag;
    if (TRB) {
      // B stored [N,K] row-major: contiguous along K -> two b128 loads
      const v8bf* bp = (const v8bf*)(Bh + (size_t)(tn + l16) * ldb + k0 + 8 * hgrp);
      bfrag = cat8(bp[0], bp[2]);
    } else {
#pragma unroll
      for (int e = 0; e < 16; ++e) {
        int kk = k0 + (e & 7) + 8 * hgrp + 16 * (e >> 3);
        size_t bi = (size_t)kk * ldb + (tn + l16);
        bfrag[e] = BF32 ? (__bf16)Bf[bi] : Bh[bi];
      }
    }
#pragma unroll
    for (int t = 0; t < 4; ++t) {
      v16bf afrag;
      if (TRA) {
#pragma unroll
        for (int e = 0; e < 16; ++e) {
          int kk = k0 + (e & 7) + 8 * hgrp + 16 * (e >> 3);
          afrag[e] = A[(size_t)kk * lda + (tm0 + 16 * t + l16)];
        }
      } else {
        const v8bf* ap =
            (const v8bf*)(A + (size_t)(tm0 + 16 * t + l16) * lda + k0 + 8 * hgrp);
        afrag = cat8(ap[0], ap[2]);
      }
      acc[t] = wmma_bf16(afrag, bfrag, acc[t]);
    }
  }

#pragma unroll
  for (int t = 0; t < 4; ++t)
#pragma unroll
    for (int r = 0; r < 8; ++r) {
      int row = tm0 + 16 * t + r + 8 * hgrp;
      int col = tn + l16;
      float v = acc[t][r] + (bias ? bias[col] : 0.0f);
      if (relu) v = fmaxf(v, 0.0f);
      size_t ci = (size_t)row * N + col;
      if (Cf) Cf[ci] = v;
      if (Cb) Cb[ci] = (__bf16)v;
    }
}

// ---------------------------------------------------------------------------
// Fused soft-adjacency multi-head attention (flash style, online softmax).
// Q/K/V: bf16 [N, 512] (head h = cols h*64..h*64+63). adj fp32 [N,N].
// O: bf16 [N, 512]. Grid: (N/16 query tiles, 8 heads), 1 wave / block.
// softmax( (QK^T / 8) * (0.5*adj + 0.5) ) @ V
// ---------------------------------------------------------------------------
__global__ __launch_bounds__(32, 1)
void k_attn(const __bf16* __restrict__ Q, const __bf16* __restrict__ K,
            const __bf16* __restrict__ V, const float* __restrict__ adj,
            __bf16* __restrict__ O, int N) {
  const int h    = blockIdx.y;
  const int q0   = blockIdx.x * 16;
  const int lane = threadIdx.x;
  const int hgrp = lane >> 4;
  const int l16  = lane & 15;
  __shared__ __bf16 pbuf[16 * 32];   // P tile staging (C-layout -> A-layout)

  // Q A-fragments (contraction 0..31 and 32..63): contiguous per lane
  v16bf aq0, aq1;
  {
    const v8bf* qp =
        (const v8bf*)(Q + (size_t)(q0 + l16) * NDH + h * 64 + 8 * hgrp);
    aq0 = cat8(qp[0], qp[2]);
    aq1 = cat8(qp[4], qp[6]);
  }

  v8f o0, o1, o2, o3;
  float mrow[8], lrow[8];
#pragma unroll
  for (int r = 0; r < 8; ++r) {
    o0[r] = o1[r] = o2[r] = o3[r] = 0.0f;
    mrow[r] = -1e30f;
    lrow[r] = 0.0f;
  }

  for (int kt = 0; kt < N; kt += 32) {
    // K B-fragments for the two 16-key subtiles (B[k,n] = K[key n][k])
    v16bf bk0a, bk0b, bk1a, bk1b;
    {
      const v8bf* kp0 =
          (const v8bf*)(K + (size_t)(kt + l16) * NDH + h * 64 + 8 * hgrp);
      const v8bf* kp1 =
          (const v8bf*)(K + (size_t)(kt + 16 + l16) * NDH + h * 64 + 8 * hgrp);
      bk0a = cat8(kp0[0], kp0[2]);  bk0b = cat8(kp0[4], kp0[6]);
      bk1a = cat8(kp1[0], kp1[2]);  bk1b = cat8(kp1[4], kp1[6]);
    }
    v8f s0, s1;
#pragma unroll
    for (int r = 0; r < 8; ++r) { s0[r] = 0.0f; s1[r] = 0.0f; }
    s0 = wmma_bf16(aq0, bk0a, s0);  s0 = wmma_bf16(aq1, bk0b, s0);
    s1 = wmma_bf16(aq0, bk1a, s1);  s1 = wmma_bf16(aq1, bk1b, s1);

    // mask, scale, online softmax; store P tile directly to LDS
    // (row = r + 8*hgrp, col = l16). Reductions span 16 lanes -> DPP16.
#pragma unroll
    for (int r = 0; r < 8; ++r) {
      int row = q0 + r + 8 * hgrp;
      const float* arow = adj + (size_t)row * N + kt;
      float sc0 = s0[r] * 0.125f * (0.5f * arow[l16] + 0.5f);
      float sc1 = s1[r] * 0.125f * (0.5f * arow[16 + l16] + 0.5f);
      float mnew  = fmaxf(mrow[r], rmax16(fmaxf(sc0, sc1)));
      float scale = __expf(mrow[r] - mnew);
      mrow[r] = mnew;
      float e0 = __expf(sc0 - mnew);
      float e1 = __expf(sc1 - mnew);
      lrow[r] = lrow[r] * scale + rsum16(e0 + e1);
      o0[r] *= scale; o1[r] *= scale; o2[r] *= scale; o3[r] *= scale;
      int prow = r + 8 * hgrp;
      pbuf[prow * 32 + l16]      = (__bf16)e0;
      pbuf[prow * 32 + 16 + l16] = (__bf16)e1;
    }
    asm volatile("s_wait_dscnt 0" ::: "memory");
    v16bf pa;
    {
      const v8bf* pp = (const v8bf*)(pbuf + l16 * 32 + 8 * hgrp);
      pa = cat8(pp[0], pp[2]);
    }

    // V B-fragments (contraction over the 32 keys), 4 output column chunks
    v16bf bv0, bv1, bv2, bv3;
#pragma unroll
    for (int e = 0; e < 16; ++e) {
      int k = (e & 7) + 8 * hgrp + 16 * (e >> 3);
      size_t base = (size_t)(kt + k) * NDH + h * 64;
      bv0[e] = V[base + l16];      bv1[e] = V[base + 16 + l16];
      bv2[e] = V[base + 32 + l16]; bv3[e] = V[base + 48 + l16];
    }
    o0 = wmma_bf16(pa, bv0, o0);
    o1 = wmma_bf16(pa, bv1, o1);
    o2 = wmma_bf16(pa, bv2, o2);
    o3 = wmma_bf16(pa, bv3, o3);
  }

#pragma unroll
  for (int r = 0; r < 8; ++r) {
    int row = q0 + r + 8 * hgrp;
    float inv = 1.0f / lrow[r];
    size_t base = (size_t)row * NDH + h * 64;
    O[base + l16]      = (__bf16)(o0[r] * inv);
    O[base + 16 + l16] = (__bf16)(o1[r] * inv);
    O[base + 32 + l16] = (__bf16)(o2[r] * inv);
    O[base + 48 + l16] = (__bf16)(o3[r] * inv);
  }
}

// LayerNorm over D=64: out = LN(x + res) * g + b. One wave / row.
__global__ void k_ln(const float* __restrict__ x, const float* __restrict__ res,
                     const float* __restrict__ g, const float* __restrict__ b,
                     float* __restrict__ outf, __bf16* __restrict__ outb) {
  int row = blockIdx.x, lane = threadIdx.x;
  size_t r64 = (size_t)row * 64;
  float v0 = x[r64 + lane], v1 = x[r64 + 32 + lane];
  if (res) { v0 += res[r64 + lane]; v1 += res[r64 + 32 + lane]; }
  float s = v0 + v1;
#pragma unroll
  for (int m = 1; m < 32; m <<= 1) s += __shfl_xor(s, m, 32);
  float mean = s * (1.0f / 64.0f);
  float d0 = v0 - mean, d1 = v1 - mean;
  float q = d0 * d0 + d1 * d1;
#pragma unroll
  for (int m = 1; m < 32; m <<= 1) q += __shfl_xor(q, m, 32);
  float rstd = rsqrtf(q * (1.0f / 64.0f) + 1e-5f);
  float y0 = d0 * rstd * g[lane] + b[lane];
  float y1 = d1 * rstd * g[32 + lane] + b[32 + lane];
  outf[r64 + lane] = y0;
  outf[r64 + 32 + lane] = y1;
  if (outb) { outb[r64 + lane] = (__bf16)y0; outb[r64 + 32 + lane] = (__bf16)y1; }
}

// Gated fuse: r = sigmoid([a, up] @ Wg + bg); out = (1-r)*a + r*up
__global__ void k_gate(const float* __restrict__ a, const float* __restrict__ up,
                       const float* __restrict__ Wg, const float* __restrict__ bg,
                       float* __restrict__ outf, __bf16* __restrict__ outb) {
  int row = blockIdx.x, lane = threadIdx.x;
  size_t r64 = (size_t)row * 64;
  float a0 = a[r64 + lane], a1 = a[r64 + 32 + lane];
  float u0 = up[r64 + lane], u1 = up[r64 + 32 + lane];
  float dot = a0 * Wg[lane] + a1 * Wg[32 + lane] + u0 * Wg[64 + lane] + u1 * Wg[96 + lane];
#pragma unroll
  for (int m = 1; m < 32; m <<= 1) dot += __shfl_xor(dot, m, 32);
  float r = 1.0f / (1.0f + __expf(-(dot + bg[0])));
  float o0 = (1.0f - r) * a0 + r * u0;
  float o1 = (1.0f - r) * a1 + r * u1;
  outf[r64 + lane] = o0;
  outf[r64 + 32 + lane] = o1;
  if (outb) { outb[r64 + lane] = (__bf16)o0; outb[r64 + 32 + lane] = (__bf16)o1; }
}

// ---------------------------------------------------------------------------
// Host-side orchestration
// ---------------------------------------------------------------------------
struct AttnP {
  const float *Wq, *bq, *Wk, *bk, *Wv, *bv, *Wo, *bo;
  const float *W1, *c1, *W2, *c2, *g1, *be1, *g2, *be2;
};
struct LayerP {
  AttnP a;
  const float *W1, *c1, *W2, *c2, *g1, *be1, *g2, *be2;
};
struct Scratch {
  __bf16 *Wq_b, *Wk_b, *Wv_b, *Wo_b, *aW1_b, *aW2_b, *lW1_b, *lW2_b;
  __bf16 *qb, *kb, *vb, *ob, *x1b, *ffb, *x2b;
  float  *projf, *x1f, *fff, *af, *x2f;
};

static void launch_gemm(const void* A, int lda, int tra,
                        const void* B, int bF32, int ldb, int trb,
                        const float* bias, int relu,
                        float* Cf, __bf16* Cb, int M, int Ncols, int K,
                        hipStream_t st) {
  dim3 grid(Ncols / 16, M / 64);
  const __bf16* Ab = (const __bf16*)A;
  if (bF32) {
    k_gemm_t<true, false, false><<<grid, 32, 0, st>>>(Ab, B, bias, Cf, Cb,
                                                      M, Ncols, K, lda, ldb, relu);
  } else if (tra) {
    k_gemm_t<false, true, false><<<grid, 32, 0, st>>>(Ab, B, bias, Cf, Cb,
                                                      M, Ncols, K, lda, ldb, relu);
  } else if (trb) {
    k_gemm_t<false, false, true><<<grid, 32, 0, st>>>(Ab, B, bias, Cf, Cb,
                                                      M, Ncols, K, lda, ldb, relu);
  } else {
    k_gemm_t<false, false, false><<<grid, 32, 0, st>>>(Ab, B, bias, Cf, Cb,
                                                       M, Ncols, K, lda, ldb, relu);
  }
}

static void run_layer(int N, const float* xf, const __bf16* xb, const float* adj,
                      const LayerP& p, float* outf, __bf16* outb,
                      const Scratch& s, hipStream_t st) {
  auto cv = [&](const float* src, __bf16* dst, int n) {
    k_f2b<<<(n + 255) / 256, 256, 0, st>>>(src, dst, n);
  };
  cv(p.a.Wq, s.Wq_b, ND * NDH);
  cv(p.a.Wk, s.Wk_b, ND * NDH);
  cv(p.a.Wv, s.Wv_b, ND * NDH);
  cv(p.a.Wo, s.Wo_b, NDH * ND);
  cv(p.a.W1, s.aW1_b, ND * ND);
  cv(p.a.W2, s.aW2_b, ND * ND);
  cv(p.W1, s.lW1_b, ND * ND);
  cv(p.W2, s.lW2_b, ND * ND);

  // q,k,v projections -> bf16
  launch_gemm(xb, ND, 0, s.Wq_b, 0, NDH, 0, p.a.bq, 0, nullptr, s.qb, N, NDH, ND, st);
  launch_gemm(xb, ND, 0, s.Wk_b, 0, NDH, 0, p.a.bk, 0, nullptr, s.kb, N, NDH, ND, st);
  launch_gemm(xb, ND, 0, s.Wv_b, 0, NDH, 0, p.a.bv, 0, nullptr, s.vb, N, NDH, ND, st);

  // fused soft-adjacency attention
  dim3 ag(N / 16, NH);
  k_attn<<<ag, 32, 0, st>>>(s.qb, s.kb, s.vb, adj, s.ob, N);

  // output projection + residual LN (attn block)
  launch_gemm(s.ob, NDH, 0, s.Wo_b, 0, ND, 0, p.a.bo, 0, s.projf, nullptr, N, ND, NDH, st);
  k_ln<<<N, 32, 0, st>>>(xf, s.projf, p.a.g1, p.a.be1, s.x1f, s.x1b);

  // attn-internal FFN + LN
  launch_gemm(s.x1b, ND, 0, s.aW1_b, 0, ND, 0, p.a.c1, 1, nullptr, s.ffb, N, ND, ND, st);
  launch_gemm(s.ffb, ND, 0, s.aW2_b, 0, ND, 0, p.a.c2, 0, s.fff, nullptr, N, ND, ND, st);
  k_ln<<<N, 32, 0, st>>>(s.x1f, s.fff, p.a.g2, p.a.be2, s.af, (__bf16*)nullptr);

  // layer residual LN + layer FFN + final LN
  k_ln<<<N, 32, 0, st>>>(xf, s.af, p.g1, p.be1, s.x2f, s.x2b);
  launch_gemm(s.x2b, ND, 0, s.lW1_b, 0, ND, 0, p.c1, 1, nullptr, s.ffb, N, ND, ND, st);
  launch_gemm(s.ffb, ND, 0, s.lW2_b, 0, ND, 0, p.c2, 0, s.fff, nullptr, N, ND, ND, st);
  k_ln<<<N, 32, 0, st>>>(s.x2f, s.fff, p.g2, p.be2, outf, outb);
}

extern "C" void kernel_launch(void* const* d_in, const int* in_sizes, int n_in,
                              void* d_out, int out_size, void* d_ws, size_t ws_size,
                              hipStream_t stream) {
  (void)in_sizes; (void)n_in; (void)out_size; (void)ws_size;

  const float* feat   = (const float*)d_in[0];
  const float* rpe    = (const float*)d_in[1];
  const float* segadj = (const float*)d_in[2];
  const float* s2l    = (const float*)d_in[3];
  const float* l2r    = (const float*)d_in[4];

  int idx = 5;
  auto next = [&]() -> const float* { return (const float*)d_in[idx++]; };
  auto readAttn = [&](AttnP& a) {
    a.Wq = next(); a.bq = next(); a.Wk = next(); a.bk = next();
    a.Wv = next(); a.bv = next(); a.Wo = next(); a.bo = next();
    a.W1 = next(); a.c1 = next(); a.W2 = next(); a.c2 = next();
    a.g1 = next(); a.be1 = next(); a.g2 = next(); a.be2 = next();
  };
  auto readLayer = [&](LayerP& L) {
    readAttn(L.a);
    L.W1 = next(); L.c1 = next(); L.W2 = next(); L.c2 = next();
    L.g1 = next(); L.be1 = next(); L.g2 = next(); L.be2 = next();
  };
  LayerP region, locality, segment;
  readLayer(region); readLayer(locality); readLayer(segment);
  const float* Wgl = next(); const float* bgl = next();
  const float* Wgs = next(); const float* bgs = next();

  // Workspace bump allocator
  char* base = (char*)d_ws;
  size_t off = 0;
  auto alloc = [&](size_t bytes) -> void* {
    void* p = (void*)(base + off);
    off += (bytes + 255) & ~(size_t)255;
    return p;
  };

  __bf16* s2l_b    = (__bf16*)alloc((size_t)NLOC * NSEG * 2);
  __bf16* l2r_b    = (__bf16*)alloc((size_t)NREG * NLOC * 2);
  __bf16* T1b      = (__bf16*)alloc((size_t)NLOC * NSEG * 2);
  float*  locadj_f = (float*) alloc((size_t)NLOC * NLOC * 4);
  __bf16* locadj_b = (__bf16*)alloc((size_t)NLOC * NLOC * 2);
  __bf16* T2b      = (__bf16*)alloc((size_t)NREG * NLOC * 2);
  float*  regadj_f = (float*) alloc((size_t)NREG * NREG * 4);
  float*  seg_f    = (float*) alloc((size_t)NSEG * ND * 4);
  __bf16* seg_b    = (__bf16*)alloc((size_t)NSEG * ND * 2);
  float*  loc_f    = (float*) alloc((size_t)NLOC * ND * 4);
  __bf16* loc_b    = (__bf16*)alloc((size_t)NLOC * ND * 2);
  float*  reg_f    = (float*) alloc((size_t)NREG * ND * 4);
  __bf16* reg_b    = (__bf16*)alloc((size_t)NREG * ND * 2);
  float*  regu_f   = (float*) alloc((size_t)NREG * ND * 4);
  __bf16* regu_b   = (__bf16*)alloc((size_t)NREG * ND * 2);
  float*  locup_f  = (float*) alloc((size_t)NLOC * ND * 4);
  float*  loc2_f   = (float*) alloc((size_t)NLOC * ND * 4);
  __bf16* loc2_b   = (__bf16*)alloc((size_t)NLOC * ND * 2);
  float*  locu_f   = (float*) alloc((size_t)NLOC * ND * 4);
  __bf16* locu_b   = (__bf16*)alloc((size_t)NLOC * ND * 2);
  float*  segup_f  = (float*) alloc((size_t)NSEG * ND * 4);
  float*  seg2_f   = (float*) alloc((size_t)NSEG * ND * 4);
  __bf16* seg2_b   = (__bf16*)alloc((size_t)NSEG * ND * 2);

  Scratch s;
  s.Wq_b  = (__bf16*)alloc(ND * NDH * 2);
  s.Wk_b  = (__bf16*)alloc(ND * NDH * 2);
  s.Wv_b  = (__bf16*)alloc(ND * NDH * 2);
  s.Wo_b  = (__bf16*)alloc(NDH * ND * 2);
  s.aW1_b = (__bf16*)alloc(ND * ND * 2);
  s.aW2_b = (__bf16*)alloc(ND * ND * 2);
  s.lW1_b = (__bf16*)alloc(ND * ND * 2);
  s.lW2_b = (__bf16*)alloc(ND * ND * 2);
  s.qb    = (__bf16*)alloc((size_t)NSEG * NDH * 2);
  s.kb    = (__bf16*)alloc((size_t)NSEG * NDH * 2);
  s.vb    = (__bf16*)alloc((size_t)NSEG * NDH * 2);
  s.ob    = (__bf16*)alloc((size_t)NSEG * NDH * 2);
  s.x1b   = (__bf16*)alloc((size_t)NSEG * ND * 2);
  s.ffb   = (__bf16*)alloc((size_t)NSEG * ND * 2);
  s.x2b   = (__bf16*)alloc((size_t)NSEG * ND * 2);
  s.projf = (float*) alloc((size_t)NSEG * ND * 4);
  s.x1f   = (float*) alloc((size_t)NSEG * ND * 4);
  s.fff   = (float*) alloc((size_t)NSEG * ND * 4);
  s.af    = (float*) alloc((size_t)NSEG * ND * 4);
  s.x2f   = (float*) alloc((size_t)NSEG * ND * 4);

  // ---- pipeline ----
  k_f2b<<<(NLOC * NSEG + 255) / 256, 256, 0, stream>>>(s2l, s2l_b, NLOC * NSEG);
  k_f2b<<<(NREG * NLOC + 255) / 256, 256, 0, stream>>>(l2r, l2r_b, NREG * NLOC);
  k_addf<<<(NSEG * ND + 255) / 256, 256, 0, stream>>>(feat, rpe, seg_f, seg_b, NSEG * ND);

  // adjacency coarsening: loc_adj = s2l @ seg_adj @ s2l^T ; reg_adj likewise
  launch_gemm(s2l_b, NSEG, 0, segadj, 1, NSEG, 0, nullptr, 0,
              nullptr, T1b, NLOC, NSEG, NSEG, stream);
  launch_gemm(T1b, NSEG, 0, s2l_b, 0, NSEG, 1, nullptr, 0,
              locadj_f, locadj_b, NLOC, NLOC, NSEG, stream);
  launch_gemm(l2r_b, NLOC, 0, locadj_b, 0, NLOC, 0, nullptr, 0,
              nullptr, T2b, NREG, NLOC, NLOC, stream);
  launch_gemm(T2b, NLOC, 0, l2r_b, 0, NLOC, 1, nullptr, 0,
              regadj_f, nullptr, NREG, NREG, NLOC, stream);

  // feature pooling
  launch_gemm(s2l_b, NSEG, 0, seg_b, 0, ND, 0, nullptr, 0,
              loc_f, loc_b, NLOC, ND, NSEG, stream);
  launch_gemm(l2r_b, NLOC, 0, loc_b, 0, ND, 0, nullptr, 0,
              reg_f, reg_b, NREG, ND, NLOC, stream);

  // region transformer, upsample, gate
  run_layer(NREG, reg_f, reg_b, regadj_f, region, regu_f, regu_b, s, stream);
  launch_gemm(l2r_b, NLOC, 1, regu_b, 0, ND, 0, nullptr, 0,
              locup_f, nullptr, NLOC, ND, NREG, stream);
  k_gate<<<NLOC, 32, 0, stream>>>(loc_f, locup_f, Wgl, bgl, loc2_f, loc2_b);

  // locality transformer, upsample, gate
  run_layer(NLOC, loc2_f, loc2_b, locadj_f, locality, locu_f, locu_b, s, stream);
  launch_gemm(s2l_b, NSEG, 1, locu_b, 0, ND, 0, nullptr, 0,
              segup_f, nullptr, NSEG, ND, NLOC, stream);
  k_gate<<<NSEG, 32, 0, stream>>>(seg_f, segup_f, Wgs, bgs, seg2_f, seg2_b);

  // segment transformer -> final output
  run_layer(NSEG, seg2_f, seg2_b, segadj, segment, (float*)d_out, (__bf16*)nullptr,
            s, stream);
}
